// CrossModel_55499567399318
// MI455X (gfx1250) — matrix operand order repaired and checked by reading, not compiled, not run on hardware
//
#include <hip/hip_runtime.h>
#include <hip/hip_bf16.h>

typedef __attribute__((ext_vector_type(16))) _Float16 v16h;
typedef __attribute__((ext_vector_type(8)))  _Float16 v8h;
typedef __attribute__((ext_vector_type(4)))  _Float16 v4h;
typedef __attribute__((ext_vector_type(8)))  float    v8f;

#define TPB 256
#define WAVES_PER_BLOCK 8

static inline int cdiv(long a, long b) { return (int)((a + b - 1) / b); }

// ---------------- degree / normalization ----------------

__global__ __launch_bounds__(TPB) void k_init_one(float* deg, int n) {
    int i = blockIdx.x * blockDim.x + threadIdx.x;
    if (i < n) deg[i] = 1.0f;  // self-loop weight
}

__global__ __launch_bounds__(TPB) void k_deg_accum(const int* __restrict__ dst,
                                                   const float* __restrict__ w,
                                                   float* deg, int E) {
    int i = blockIdx.x * blockDim.x + threadIdx.x;
    if (i < E) atomicAdd(&deg[dst[i]], w[i]);
}

__global__ __launch_bounds__(TPB) void k_rsqrt_ip(float* deg, int n) {
    int i = blockIdx.x * blockDim.x + threadIdx.x;
    if (i < n) {
        float d = deg[i];
        deg[i] = (d > 0.0f) ? rsqrtf(d) : 0.0f;   // becomes dis[]
    }
}

// ---------------- dtype conversion ----------------

__global__ __launch_bounds__(TPB) void k_cvt_f32_f16_x4(const float* __restrict__ s,
                                                        _Float16* __restrict__ d, int n4) {
    int i = blockIdx.x * blockDim.x + threadIdx.x;
    if (i >= n4) return;
    float4 v = ((const float4*)s)[i];
    v4h h; h[0] = (_Float16)v.x; h[1] = (_Float16)v.y; h[2] = (_Float16)v.z; h[3] = (_Float16)v.w;
    ((v4h*)d)[i] = h;
}

// W [K x Nout] f32  ->  Wt [Nout x K] f16 (so B fragments are contiguous per column)
__global__ __launch_bounds__(TPB) void k_cvtT(const float* __restrict__ s,
                                              _Float16* __restrict__ d, int K, int Nout) {
    int i = blockIdx.x * blockDim.x + threadIdx.x;
    if (i >= K * Nout) return;
    int k = i / Nout, n = i % Nout;
    d[n * K + k] = (_Float16)s[i];
}

// ---------------- WMMA GEMM:  C[M x Nout] = A[M x 128] @ Bt[Nout x 128]^T ----------------
// Each wave owns a 16 x (16*T) strip: A fragment loaded once per k-block and
// reused across T column tiles (T wmma's per A load).

template <int T>
__global__ __launch_bounds__(TPB) void k_gemm_f16_wmma(const _Float16* __restrict__ A,
                                                       const _Float16* __restrict__ Bt,
                                                       float* __restrict__ C,
                                                       int M, int Nout) {
    const int K = 128;
    const int lane = threadIdx.x & 31;
    const int wave = threadIdx.x >> 5;
    const int stripsN = (Nout >> 4) / T;
    const int total = (M >> 4) * stripsN;
    const int gw = blockIdx.x * WAVES_PER_BLOCK + wave;
    if (gw >= total) return;                 // wave-uniform: EXEC stays all-ones
    const int tm = gw / stripsN, sn = gw % stripsN;
    const int laneLo = lane & 15, laneHi = lane >> 4;

    const _Float16* arow  = A + (size_t)(tm * 16 + laneLo) * K;            // A row M = lane%16
    const _Float16* bbase = Bt + (size_t)(sn * T * 16 + laneLo) * K;       // B col N = lane%16

    v8f acc[T];
#pragma unroll
    for (int t = 0; t < T; ++t) acc[t] = (v8f){};

#pragma unroll
    for (int kb = 0; kb < 4; ++kb) {
        // A 16x32 frag: 16 contiguous K starting at 16*laneHi (elems 0-7, then 8-15)
        const int abase = kb * 32 + 16 * laneHi;
        v8h a0 = *(const v8h*)(arow + abase);
        v8h a1 = *(const v8h*)(arow + abase + 8);
        v16h af;
#pragma unroll
        for (int i = 0; i < 8; ++i) { af[i] = a0[i]; af[i + 8] = a1[i]; }
#pragma unroll
        for (int t = 0; t < T; ++t) {
            // B 32x16 frag: 16 contiguous K starting at 16*laneHi, column tile t
            v16h bf = *(const v16h*)(bbase + (size_t)t * 16 * K + kb * 32 + 16 * laneHi);
            acc[t] = __builtin_amdgcn_wmma_f32_16x16x32_f16(false, af, false, bf,
                                                            (short)0, acc[t], false, false);
        }
    }
    // C 16x16 f32: elem r -> row tm*16 + r + 8*laneHi, col (sn*T+t)*16 + laneLo
#pragma unroll
    for (int t = 0; t < T; ++t) {
        float* crow = C + (size_t)(tm * 16 + 8 * laneHi) * Nout + (sn * T + t) * 16 + laneLo;
#pragma unroll
        for (int r = 0; r < 8; ++r) crow[(size_t)r * Nout] = acc[t][r];
    }
}

// ---------------- aggregation ----------------

// hout[n,f] = hpre[n,f] * dis[n]^2   (self-loop term, also initializes hout)
__global__ __launch_bounds__(TPB) void k_self_term(const float* __restrict__ hpre,
                                                   const float* __restrict__ dis,
                                                   float* __restrict__ hout,
                                                   int total4, int Fc) {
    int i = blockIdx.x * blockDim.x + threadIdx.x;
    if (i >= total4) return;
    int n = i / Fc;
    float sn = dis[n] * dis[n];
    float4 v = ((const float4*)hpre)[i];
    v.x *= sn; v.y *= sn; v.z *= sn; v.w *= sn;
    ((float4*)hout)[i] = v;
}

// hout[dst,f] += dis[src]*w*dis[dst] * hpre[src,f]; thread = (edge, 4-feature chunk)
__global__ __launch_bounds__(TPB) void k_edge_scatter(const int* __restrict__ src,
                                                      const int* __restrict__ dst,
                                                      const float* __restrict__ w,
                                                      const float* __restrict__ dis,
                                                      const float* __restrict__ hpre,
                                                      float* __restrict__ hout,
                                                      int E, int F, int Fc) {
    int idx = blockIdx.x * blockDim.x + threadIdx.x;
    int e = idx / Fc;
    if (e >= E) return;
    int c = idx - e * Fc;
    int s = src[e], d = dst[e];
    float coef = dis[s] * w[e] * dis[d];
    float4 v = *(const float4*)(hpre + (size_t)s * F + 4 * c);
    float* o = hout + (size_t)d * F + 4 * c;
    atomicAdd(o + 0, coef * v.x);
    atomicAdd(o + 1, coef * v.y);
    atomicAdd(o + 2, coef * v.z);
    atomicAdd(o + 3, coef * v.w);
}

__global__ __launch_bounds__(TPB) void k_bias_act(float* __restrict__ h,
                                                  const float* __restrict__ b,
                                                  int total4, int Fc, int relu) {
    int i = blockIdx.x * blockDim.x + threadIdx.x;
    if (i >= total4) return;
    int c = i % Fc;
    float4 v = ((float4*)h)[i];
    float4 bb = ((const float4*)b)[c];
    v.x += bb.x; v.y += bb.y; v.z += bb.z; v.w += bb.w;
    if (relu) {
        v.x = fmaxf(v.x, 0.0f); v.y = fmaxf(v.y, 0.0f);
        v.z = fmaxf(v.z, 0.0f); v.w = fmaxf(v.w, 0.0f);
    }
    ((float4*)h)[i] = v;
}

// xs1[s0[i],:] = h1[s0[i],:] + h2[s1[i],:]  (set semantics on duplicates, like JAX .set)
__global__ __launch_bounds__(TPB) void k_seed_set(const int* __restrict__ s0,
                                                  const int* __restrict__ s1,
                                                  const float* __restrict__ h1,
                                                  const float* __restrict__ h2,
                                                  _Float16* __restrict__ xs1,
                                                  _Float16* __restrict__ xs2,
                                                  int S, int F) {
    int idx = blockIdx.x * blockDim.x + threadIdx.x;
    if (idx >= S * F) return;
    int i = idx / F, f = idx - i * F;
    int a = s0[i], b = s1[i];
    float va = h1[(size_t)a * F + f];
    float vb = h2[(size_t)b * F + f];
    xs1[(size_t)a * F + f] = (_Float16)(va + vb);
    xs2[(size_t)b * F + f] = (_Float16)(vb + va);
}

// ---------------- launch ----------------

extern "C" void kernel_launch(void* const* d_in, const int* in_sizes, int n_in,
                              void* d_out, int out_size, void* d_ws, size_t ws_size,
                              hipStream_t stream) {
    const float* x1  = (const float*)d_in[0];
    const int*   ei1 = (const int*)d_in[1];
    const float* ew1 = (const float*)d_in[2];
    const float* x2  = (const float*)d_in[3];
    const int*   ei2 = (const int*)d_in[4];
    const float* ew2 = (const float*)d_in[5];
    const int*   sd  = (const int*)d_in[6];
    const float* W1  = (const float*)d_in[7];
    const float* b1  = (const float*)d_in[8];
    const float* W2  = (const float*)d_in[9];
    const float* b2  = (const float*)d_in[10];
    const float* W3  = (const float*)d_in[11];
    const float* b3  = (const float*)d_in[12];

    const int F = 128, OUT = 64;
    const int N = in_sizes[0] / F;
    const int E = in_sizes[2];
    const int S = in_sizes[6] / 2;
    const int* src1 = ei1, *dst1 = ei1 + E;
    const int* src2 = ei2, *dst2 = ei2 + E;
    const int* s0 = sd, *s1 = sd + S;

    // workspace carve-out
    char* p = (char*)d_ws;
    float* dis1  = (float*)p; p += (size_t)N * 4;                 // deg -> dis (in place)
    float* dis2  = (float*)p; p += (size_t)N * 4;
    float* h1pre = (float*)p; p += (size_t)N * F * 4;             // GEMM1 out, later t1
    float* h2pre = (float*)p; p += (size_t)N * F * 4;             //              later t2
    float* h1    = (float*)p; p += (size_t)N * F * 4;             // aggregated layer 1
    float* h2    = (float*)p; p += (size_t)N * F * 4;
    _Float16* x1h = (_Float16*)p; p += (size_t)N * F * 2;         // f16 x, later xs1h
    _Float16* x2h = (_Float16*)p; p += (size_t)N * F * 2;
    _Float16* w1t = (_Float16*)p; p += (size_t)F * F * 2;         // transposed f16 weights
    _Float16* w2t = (_Float16*)p; p += (size_t)F * F * 2;
    _Float16* w3t = (_Float16*)p; p += (size_t)F * OUT * 2;

    float* o1 = (float*)d_out;
    float* o2 = o1 + (size_t)N * OUT;

    // 1) degrees -> dis = rsqrt(deg)
    k_init_one<<<cdiv(2 * N, TPB), TPB, 0, stream>>>(dis1, 2 * N);   // dis1,dis2 contiguous
    k_deg_accum<<<cdiv(E, TPB), TPB, 0, stream>>>(dst1, ew1, dis1, E);
    k_deg_accum<<<cdiv(E, TPB), TPB, 0, stream>>>(dst2, ew2, dis2, E);
    k_rsqrt_ip<<<cdiv(2 * N, TPB), TPB, 0, stream>>>(dis1, 2 * N);

    // 2) convert inputs/weights to f16 (weights transposed for contiguous B frags)
    k_cvt_f32_f16_x4<<<cdiv((long)N * F / 4, TPB), TPB, 0, stream>>>(x1, x1h, N * F / 4);
    k_cvt_f32_f16_x4<<<cdiv((long)N * F / 4, TPB), TPB, 0, stream>>>(x2, x2h, N * F / 4);
    k_cvtT<<<cdiv(F * F, TPB), TPB, 0, stream>>>(W1, w1t, F, F);
    k_cvtT<<<cdiv(F * F, TPB), TPB, 0, stream>>>(W2, w2t, F, F);
    k_cvtT<<<cdiv(F * OUT, TPB), TPB, 0, stream>>>(W3, w3t, F, OUT);

    // 3) layer-1 GEMMs (WMMA, 16x64 strips): h_pre = x @ W
    {
        int strips = (N / 16) * ((F / 16) / 4);
        int blocks = cdiv(strips, WAVES_PER_BLOCK);
        k_gemm_f16_wmma<4><<<blocks, TPB, 0, stream>>>(x1h, w1t, h1pre, N, F);
        k_gemm_f16_wmma<4><<<blocks, TPB, 0, stream>>>(x2h, w2t, h2pre, N, F);
    }

    // 4) layer-1 aggregation: self term, edge scatter, bias+relu
    int nf4 = N * F / 4, fc = F / 4;
    k_self_term<<<cdiv(nf4, TPB), TPB, 0, stream>>>(h1pre, dis1, h1, nf4, fc);
    k_self_term<<<cdiv(nf4, TPB), TPB, 0, stream>>>(h2pre, dis2, h2, nf4, fc);
    k_edge_scatter<<<cdiv((long)E * fc, TPB), TPB, 0, stream>>>(src1, dst1, ew1, dis1, h1pre, h1, E, F, fc);
    k_edge_scatter<<<cdiv((long)E * fc, TPB), TPB, 0, stream>>>(src2, dst2, ew2, dis2, h2pre, h2, E, F, fc);
    k_bias_act<<<cdiv(nf4, TPB), TPB, 0, stream>>>(h1, b1, nf4, fc, 1);
    k_bias_act<<<cdiv(nf4, TPB), TPB, 0, stream>>>(h2, b2, nf4, fc, 1);

    // 5) xs = h + cross-seed rows, stored as f16 (reuse x1h/x2h buffers)
    k_cvt_f32_f16_x4<<<cdiv(nf4, TPB), TPB, 0, stream>>>(h1, x1h, nf4);
    k_cvt_f32_f16_x4<<<cdiv(nf4, TPB), TPB, 0, stream>>>(h2, x2h, nf4);
    k_seed_set<<<cdiv((long)S * F, TPB), TPB, 0, stream>>>(s0, s1, h1, h2, x1h, x2h, S, F);

    // 6) layer-2 GEMMs (WMMA, 16x64 strips): t = xs @ W3  (reuse h1pre/h2pre as t1/t2)
    {
        int strips = (N / 16) * ((OUT / 16) / 4);
        int blocks = cdiv(strips, WAVES_PER_BLOCK);
        k_gemm_f16_wmma<4><<<blocks, TPB, 0, stream>>>(x1h, w3t, h1pre, N, OUT);
        k_gemm_f16_wmma<4><<<blocks, TPB, 0, stream>>>(x2h, w3t, h2pre, N, OUT);
    }

    // 7) layer-2 aggregation directly into d_out, then + b3 (no relu)
    int no4 = N * OUT / 4, oc = OUT / 4;
    k_self_term<<<cdiv(no4, TPB), TPB, 0, stream>>>(h1pre, dis1, o1, no4, oc);
    k_self_term<<<cdiv(no4, TPB), TPB, 0, stream>>>(h2pre, dis2, o2, no4, oc);
    k_edge_scatter<<<cdiv((long)E * oc, TPB), TPB, 0, stream>>>(src1, dst1, ew1, dis1, h1pre, o1, E, OUT, oc);
    k_edge_scatter<<<cdiv((long)E * oc, TPB), TPB, 0, stream>>>(src2, dst2, ew2, dis2, h2pre, o2, E, OUT, oc);
    k_bias_act<<<cdiv(no4, TPB), TPB, 0, stream>>>(o1, b3, no4, oc, 0);
    k_bias_act<<<cdiv(no4, TPB), TPB, 0, stream>>>(o2, b3, no4, oc, 0);
}
